// TopSimilarTokens_32418413150670
// MI455X (gfx1250) — matrix operand order repaired and checked by reading, not compiled, not run on hardware
//
#include <hip/hip_runtime.h>
#include <hip/hip_bf16.h>
#include <math.h>

// ---------------------------------------------------------------------------
// TopSimilarTokens for MI455X (gfx1250, wave32, WMMA)
//   x:(2048,768) f32, E:(50000,768) f32, k=16
//   out[b][j][:] = E[idx[j*128 + b/16][b%16]][:], idx = per-row top-16 of
//   cosine similarity, sorted descending (matches jax.lax.top_k order).
// GEMM path: bf16 inputs, f32 accumulate via v_wmma_f32_16x16x32_bf16.
// E tiles are staged with CDNA5 async DMA: global_load_async_to_lds_b128
// (ASYNCcnt / s_wait_asynccnt), ping-pong double buffered in LDS.
// ---------------------------------------------------------------------------

typedef __bf16 bf16_t;
typedef bf16_t v16bf __attribute__((ext_vector_type(16)));
typedef bf16_t v8bf  __attribute__((ext_vector_type(8)));
typedef float  v8f   __attribute__((ext_vector_type(8)));

#define B_ROWS 2048
#define N_EMB  50000
#define C_DIM  768
#define TOPK   16
#define SEG    2048
#define NSEG   25          // ceil(50000 / 2048)
#define EPSV   1e-8f

#define A_STRIDE 776       // 768 + 8 bf16 pad (bank-conflict-free b128 frags)
#define E_STRIDE 40        // 32 + 8 bf16 pad
#define S_STRIDE 132       // 128 + 4 f32 pad

// --------------------------- helpers ---------------------------------------

__device__ __forceinline__ v16bf load_frag_split(const bf16_t* p0, const bf16_t* p1) {
    v8bf lo = *(const v8bf*)p0;
    v8bf hi = *(const v8bf*)p1;
    return __builtin_shufflevector(lo, hi, 0, 1, 2, 3, 4, 5, 6, 7,
                                           8, 9, 10, 11, 12, 13, 14, 15);
}

// CDNA5 async DMA: 32 contiguous bytes per lane, global -> LDS, no VGPR
// round-trip. Per ISA 10.4: the instruction offset is added to BOTH the LDS
// and the global address, so one address pair serves both 16-byte beats.
__device__ __forceinline__ void async_stage32(uint32_t lds_addr, const bf16_t* gsrc) {
    asm volatile(
        "global_load_async_to_lds_b128 %0, %1, off\n\t"
        "global_load_async_to_lds_b128 %0, %1, off offset:16"
        :
        : "v"(lds_addr), "v"(gsrc)
        : "memory");
}

__device__ __forceinline__ void wait_async0() {
    asm volatile("s_wait_asynccnt 0x0" ::: "memory");
}

// --------------------- 1) f32 -> bf16 + row norms ---------------------------

__global__ __launch_bounds__(256)
void convert_rows(const float* __restrict__ src, bf16_t* __restrict__ dst,
                  float* __restrict__ norms) {
    const int row = blockIdx.x;
    const float* s = src + (size_t)row * C_DIM;
    bf16_t*      d = dst + (size_t)row * C_DIM;
    float ss = 0.f;
    for (int c = threadIdx.x; c < C_DIM; c += 256) {
        float v = s[c];
        ss += v * v;
        d[c] = (bf16_t)v;
    }
    __shared__ float red[256];
    red[threadIdx.x] = ss;
    __syncthreads();
    for (int o = 128; o > 0; o >>= 1) {
        if (threadIdx.x < o) red[threadIdx.x] += red[threadIdx.x + o];
        __syncthreads();
    }
    if (threadIdx.x == 0) norms[row] = sqrtf(red[0]);
}

// ------------- 2) fused bf16 WMMA GEMM + per-segment top-16 -----------------
// grid = (NSEG, 128); block = 256 threads = 8 waves.
// Each block: 16 rows of x  x  one 2048-column segment of E.

__global__ __launch_bounds__(256)
void gemm_topk(const bf16_t* __restrict__ xb, const bf16_t* __restrict__ eb,
               const float* __restrict__ xn, const float* __restrict__ en,
               float* __restrict__ candV, int* __restrict__ candI) {
    const int seg   = blockIdx.x;       // 0..NSEG-1
    const int mbase = blockIdx.y * 16;  // 0..2032
    const int tid   = threadIdx.x;
    const int lane  = tid & 31;
    const int wv    = tid >> 5;         // 0..7

    __shared__ bf16_t As[16 * A_STRIDE];        // 24.8 KB, resident A tile
    __shared__ bf16_t Es[2][128 * E_STRIDE];    // 20.5 KB, double-buffered B
    __shared__ float  simsS[16 * S_STRIDE];     // 8.4 KB
    __shared__ float  tvS[256];                 // running top-16 values
    __shared__ int    tiS[256];                 // running top-16 indices
    __shared__ float  xnS[16];

    tvS[tid] = -INFINITY;
    tiS[tid] = 0x7fffffff;
    if (tid < 16) xnS[tid] = xn[mbase + tid];

    // Load A tile (16 x 768 bf16) into LDS: 16 threads/row, 48 elems each.
    {
        const int row = tid >> 4;
        const int c0  = (tid & 15) * 48;
        const uint4* g  = (const uint4*)(xb + (size_t)(mbase + row) * C_DIM + c0);
        uint4*       dl = (uint4*)&As[row * A_STRIDE + c0];
#pragma unroll
        for (int i = 0; i < 6; ++i) dl[i] = g[i];
    }

    // E staging: thread t owns row t>>1, 32-byte half t&1 of each 32-K slab.
    const int erow  = tid >> 1;
    const int ehoff = (tid & 1) * 16;   // bf16 elements
    // LDS byte address = low 32 bits of the generic (flat) pointer.
    const uint32_t lds0 = (uint32_t)(uintptr_t)&Es[0][erow * E_STRIDE + ehoff];
    const uint32_t lds1 = (uint32_t)(uintptr_t)&Es[1][erow * E_STRIDE + ehoff];

    // WMMA fragment addressing (wave32; contiguous K order, same for A and B):
    const int arow = lane & 15;               // A: row
    const int aoff = (lane >> 4) * 16;        // A: contiguous K-16 chunk
    const int brow = wv * 16 + (lane & 15);   // B: column (E row in tile)
    const int boff = (lane >> 4) * 16;        // B: contiguous K-16 chunk

    float vmin = -INFINITY;   // per-row scanner state (threads 0..15)
    int   amin = 0;

    __syncthreads();          // A tile + top-k init visible

    for (int ni = 0; ni < SEG / 128; ++ni) {
        const int colbase = seg * SEG + ni * 128;

        int nrow = colbase + erow;
        if (nrow > N_EMB - 1) nrow = N_EMB - 1;
        const bf16_t* gbase = eb + (size_t)nrow * C_DIM + ehoff;

        // Prologue: DMA K-step 0 into buffer 0.
        async_stage32(lds0, gbase);

        v8f acc = {0.f, 0.f, 0.f, 0.f, 0.f, 0.f, 0.f, 0.f};

        // 24 K-steps of 32; async DMA of step ks+1 overlaps WMMA on step ks.
        for (int ks = 0; ks < 24; ++ks) {
            wait_async0();       // own writes to buf[ks&1] complete
            __syncthreads();     // everyone's writes complete; prev readers done
            if (ks < 23) {
                // next buffer = (ks+1)&1
                async_stage32((ks & 1) ? lds0 : lds1, gbase + (ks + 1) * 32);
            }
            const int kk = ks * 32;
            const bf16_t* ab = &As[arow * A_STRIDE + kk + aoff];
            v16bf afrag = load_frag_split(ab, ab + 8);
            v16bf bfrag = *(const v16bf*)&Es[ks & 1][brow * E_STRIDE + boff];
            acc = __builtin_amdgcn_wmma_f32_16x16x32_bf16(
                      false, afrag, false, bfrag, (short)0, acc, false, false);
        }

        // Normalize and spill the 16x128 sims tile to LDS.
        // C/D layout: VGPR j, lanes 0-15 -> row j, lanes 16-31 -> row j+8.
        const int lcol = wv * 16 + (lane & 15);
        const int ncol = colbase + lcol;
        const bool colok = (ncol < N_EMB);
        const float envl = en[colok ? ncol : (N_EMB - 1)];
#pragma unroll
        for (int j = 0; j < 8; ++j) {
            const int row = (lane < 16) ? j : (j + 8);
            float sim;
            if (colok) {
                float denom = fmaxf(xnS[row] * envl, EPSV);
                sim = acc[j] / denom;
            } else {
                sim = -INFINITY;
            }
            simsS[row * S_STRIDE + lcol] = sim;
        }
        __syncthreads();

        // Running top-16 per row: threshold test, argmin replace.
        if (tid < 16) {
            const int row = tid;
            for (int c = 0; c < 128; ++c) {
                float v = simsS[row * S_STRIDE + c];
                if (v > vmin) {
                    tvS[row * 16 + amin] = v;
                    tiS[row * 16 + amin] = colbase + c;
                    float m = tvS[row * 16];
                    int   a = 0;
#pragma unroll
                    for (int s2 = 1; s2 < 16; ++s2) {
                        float w = tvS[row * 16 + s2];
                        if (w < m) { m = w; a = s2; }
                    }
                    vmin = m; amin = a;
                }
            }
        }
        __syncthreads();
    }

    // Emit this segment's 16 candidates per row.
    {
        const int row  = tid >> 4;
        const int slot = tid & 15;
        const size_t o = ((size_t)(mbase + row) * NSEG + seg) * TOPK + slot;
        candV[o] = tvS[tid];
        candI[o] = tiS[tid];
    }
}

// ------------- 3) merge NSEG*16 candidates -> sorted top-16 -----------------

__global__ __launch_bounds__(256)
void merge_topk(const float* __restrict__ candV, const int* __restrict__ candI,
                int* __restrict__ tki) {
    __shared__ float mv[256 * 16];
    __shared__ int   mi[256 * 16];
    const int tid = threadIdx.x;
    const int b   = blockIdx.x * 256 + tid;
    float* v  = &mv[tid * 16];
    int*   ix = &mi[tid * 16];
#pragma unroll
    for (int s = 0; s < 16; ++s) { v[s] = -INFINITY; ix[s] = 0x7fffffff; }
    const size_t base = (size_t)b * NSEG * TOPK;
    for (int t = 0; t < NSEG * TOPK; ++t) {
        float cv = candV[base + t];
        int   ci = candI[base + t];
        if ((cv > v[15]) || (cv == v[15] && ci < ix[15])) {
            int p = 15;
            while (p > 0 && ((cv > v[p - 1]) ||
                             (cv == v[p - 1] && ci < ix[p - 1]))) {
                v[p] = v[p - 1]; ix[p] = ix[p - 1]; --p;
            }
            v[p] = cv; ix[p] = ci;
        }
    }
    for (int s = 0; s < 16; ++s) tki[b * 16 + s] = ix[s];
}

// ------------- 4) gather with torch interleave reshape ----------------------
// out[b][j][:] = E[ idx[j*128 + b/16][b%16] ][:]

__global__ __launch_bounds__(256)
void gather_out(const float* __restrict__ emb, const int* __restrict__ tki,
                float* __restrict__ out) {
    const int bid = blockIdx.x;      // 0 .. B*K-1
    const int b = bid >> 4;
    const int j = bid & 15;
    const int n = tki[((j * 128) + (b >> 4)) * TOPK + (b & 15)];
    const float* s = emb + (size_t)n * C_DIM;
    float*       d = out + ((size_t)b * TOPK + j) * C_DIM;
    for (int c = threadIdx.x; c < C_DIM; c += 256) d[c] = s[c];
}

// --------------------------------------------------------------------------

extern "C" void kernel_launch(void* const* d_in, const int* in_sizes, int n_in,
                              void* d_out, int out_size, void* d_ws, size_t ws_size,
                              hipStream_t stream) {
    const float* x = (const float*)d_in[0];           // (2048, 768)
    const float* e = (const float*)d_in[1];           // (50000, 768)
    // d_in[2] is k == 16 (compile-time TOPK)

    char* ws = (char*)d_ws;
    size_t off = 0;
    bf16_t* xb  = (bf16_t*)(ws + off); off += (size_t)B_ROWS * C_DIM * sizeof(bf16_t);
    bf16_t* ebf = (bf16_t*)(ws + off); off += (size_t)N_EMB * C_DIM * sizeof(bf16_t);
    float*  xn  = (float*)(ws + off);  off += (size_t)B_ROWS * sizeof(float);
    float*  en  = (float*)(ws + off);  off += (size_t)N_EMB * sizeof(float);
    off = (off + 15) & ~(size_t)15;
    float*  cV  = (float*)(ws + off);  off += (size_t)B_ROWS * NSEG * TOPK * sizeof(float);
    int*    cI  = (int*)(ws + off);    off += (size_t)B_ROWS * NSEG * TOPK * sizeof(int);
    int*    tki = (int*)(ws + off);    off += (size_t)B_ROWS * TOPK * sizeof(int);
    (void)ws_size; (void)in_sizes; (void)n_in; (void)out_size;

    convert_rows<<<B_ROWS, 256, 0, stream>>>(x, xb, xn);
    convert_rows<<<N_EMB, 256, 0, stream>>>(e, ebf, en);
    gemm_topk<<<dim3(NSEG, 128), 256, 0, stream>>>(xb, ebf, xn, en, cV, cI);
    merge_topk<<<B_ROWS / 256, 256, 0, stream>>>(cV, cI, tki);
    gather_out<<<B_ROWS * TOPK, 256, 0, stream>>>(e, tki, (float*)d_out);
}
